// FAT_S_24988119728141
// MI455X (gfx1250) — matrix operand order repaired and checked by reading, not compiled, not run on hardware
//
#include <hip/hip_runtime.h>
#include <math.h>

// Problem constants (from reference): B=8, C=512, Cq=64, H=W=96
#define B_   8
#define C_   512
#define CQ_  64
#define H_   96
#define W_   96
#define N_   (H_ * W_)    // 9216 pixels per batch
#define L_   192          // softmax row length (H + W)

typedef _Float16 half_t;
typedef half_t v8h  __attribute__((ext_vector_type(8)));
typedef half_t v16h __attribute__((ext_vector_type(16)));
typedef float  v8f  __attribute__((ext_vector_type(8)));

// D(16x16 f32) = A(16x32 f16) * B(32x16 f16) + C
__device__ __forceinline__ v8f wmma_f16(v16h a, v16h b, v8f c) {
    return __builtin_amdgcn_wmma_f32_16x16x32_f16(
        /*neg_a=*/false, a, /*neg_b=*/false, b,
        /*c_mod=*/(short)0, c, /*reuse_a=*/false, /*reuse_b=*/false);
}

// A fragment: lane holds K = k0+16*kh+{0..15} in elements 0..15 (row-major
// source, K contiguous within the row) -> single 32-byte load.
__device__ __forceinline__ v16h load_a(const half_t* __restrict__ row, int k0, int kh) {
    return *(const v16h*)(row + k0 + 16 * kh);
}

// B fragment: lane holds K = k0+16*kh+{0..15} for its column (column pointer
// has K contiguous) -> single 32-byte load.
__device__ __forceinline__ v16h load_b(const half_t* __restrict__ col, int k0, int kh) {
    return *(const v16h*)(col + k0 + 16 * kh);
}

// ---------------------------------------------------------------------------
// Elementwise f32 -> f16 (weight matrices)
// ---------------------------------------------------------------------------
__global__ __launch_bounds__(256)
void f32_to_f16_kernel(const float* __restrict__ src, half_t* __restrict__ dst, int n)
{
    const int i = blockIdx.x * 256 + threadIdx.x;
    if (i < n) dst[i] = (half_t)src[i];
}

// ---------------------------------------------------------------------------
// (B, C, N) f32  ->  (B, N, C) f16 transpose via LDS tile.
// grid = (N/32, C/32, B), block = (32, 8)
// ---------------------------------------------------------------------------
__global__ __launch_bounds__(256)
void transpose_to_f16_kernel(const float* __restrict__ src, half_t* __restrict__ dst)
{
    __shared__ float tile[32][33];
    const int b  = blockIdx.z;
    const int p0 = blockIdx.x * 32;
    const int c0 = blockIdx.y * 32;
    const int tx = threadIdx.x, ty = threadIdx.y;

    const float* s = src + ((size_t)b * C_ + c0) * N_ + p0;
#pragma unroll
    for (int j = 0; j < 4; ++j)
        tile[ty + 8 * j][tx] = s[(size_t)(ty + 8 * j) * N_ + tx];
    __syncthreads();
    half_t* d = dst + ((size_t)b * N_ + p0) * C_ + c0;
#pragma unroll
    for (int j = 0; j < 4; ++j)
        d[(size_t)(ty + 8 * j) * C_ + tx] = (half_t)tile[tx][ty + 8 * j];
}

// ---------------------------------------------------------------------------
// q/k projection: outh[b, p, m] = f16( sum_c Wh[m, c] * inh[b, p, c] + bias[m] )
// Wh (O, C) f16 row-major; inh (B, N, C) f16 pixel-major; outh (B, N, O) f16.
// 64x64 tile per wave (4x4 WMMA accumulators).
// grid = (O/64, N/64, B), block = 32
// ---------------------------------------------------------------------------
__global__ __launch_bounds__(32)
void proj_qk_kernel(const half_t* __restrict__ Wh, const float* __restrict__ bias,
                    const half_t* __restrict__ inh, half_t* __restrict__ outh, int O)
{
    const int b    = blockIdx.z;
    const int m0   = blockIdx.x * 64;
    const int p0   = blockIdx.y * 64;
    const int lane = threadIdx.x;
    const int mrow = lane & 15;
    const int kh   = lane >> 4;

    const half_t* inb = inh + (size_t)b * N_ * C_;
    const half_t* ar[4];
    const half_t* bc[4];
#pragma unroll
    for (int mi = 0; mi < 4; ++mi) ar[mi] = Wh + (size_t)(m0 + mi * 16 + mrow) * C_;
#pragma unroll
    for (int ni = 0; ni < 4; ++ni) bc[ni] = inb + (size_t)(p0 + ni * 16 + mrow) * C_;

    v8f acc[4][4] = {};
    for (int c0 = 0; c0 < C_; c0 += 32) {
        v16h a[4], bv[4];
#pragma unroll
        for (int mi = 0; mi < 4; ++mi) a[mi] = load_a(ar[mi], c0, kh);
#pragma unroll
        for (int ni = 0; ni < 4; ++ni) bv[ni] = load_b(bc[ni], c0, kh);
#pragma unroll
        for (int mi = 0; mi < 4; ++mi)
#pragma unroll
            for (int ni = 0; ni < 4; ++ni)
                acc[mi][ni] = wmma_f16(a[mi], bv[ni], acc[mi][ni]);
    }

#pragma unroll
    for (int ni = 0; ni < 4; ++ni) {
        const int p = p0 + ni * 16 + mrow;
        half_t* op = outh + ((size_t)b * N_ + p) * O + m0 + 8 * kh;
#pragma unroll
        for (int mi = 0; mi < 4; ++mi) {
            v8h pack;
#pragma unroll
            for (int i = 0; i < 8; ++i)
                pack[i] = (half_t)(acc[mi][ni][i] + bias[m0 + mi * 16 + 8 * kh + i]);
            *(v8h*)(op + mi * 16) = pack;
        }
    }
}

// ---------------------------------------------------------------------------
// v projection, dual-layout store:
//   vnat[b, c, h, w] and vtr[b, c, w, h], both f16.
// 64x64 tile per wave. grid = (C/64, N/64, B), block = 32
// ---------------------------------------------------------------------------
__global__ __launch_bounds__(32)
void proj_v_kernel(const half_t* __restrict__ Wh, const float* __restrict__ bias,
                   const half_t* __restrict__ inh,
                   half_t* __restrict__ vnat, half_t* __restrict__ vtr)
{
    const int b    = blockIdx.z;
    const int m0   = blockIdx.x * 64;
    const int p0   = blockIdx.y * 64;
    const int lane = threadIdx.x;
    const int mrow = lane & 15;
    const int kh   = lane >> 4;

    const half_t* inb = inh + (size_t)b * N_ * C_;
    const half_t* ar[4];
    const half_t* bc[4];
#pragma unroll
    for (int mi = 0; mi < 4; ++mi) ar[mi] = Wh + (size_t)(m0 + mi * 16 + mrow) * C_;
#pragma unroll
    for (int ni = 0; ni < 4; ++ni) bc[ni] = inb + (size_t)(p0 + ni * 16 + mrow) * C_;

    v8f acc[4][4] = {};
    for (int c0 = 0; c0 < C_; c0 += 32) {
        v16h a[4], bv[4];
#pragma unroll
        for (int mi = 0; mi < 4; ++mi) a[mi] = load_a(ar[mi], c0, kh);
#pragma unroll
        for (int ni = 0; ni < 4; ++ni) bv[ni] = load_b(bc[ni], c0, kh);
#pragma unroll
        for (int mi = 0; mi < 4; ++mi)
#pragma unroll
            for (int ni = 0; ni < 4; ++ni)
                acc[mi][ni] = wmma_f16(a[mi], bv[ni], acc[mi][ni]);
    }

#pragma unroll
    for (int ni = 0; ni < 4; ++ni) {
        const int p = p0 + ni * 16 + mrow;
        const int h = p / W_, w = p % W_;
#pragma unroll
        for (int mi = 0; mi < 4; ++mi) {
#pragma unroll
            for (int i = 0; i < 8; ++i) {
                const int c = m0 + mi * 16 + 8 * kh + i;
                const half_t val = (half_t)(acc[mi][ni][i] + bias[c]);
                vnat[((size_t)(b * C_ + c) * H_ + h) * W_ + w] = val;
                vtr [((size_t)(b * C_ + c) * W_ + w) * H_ + h] = val;
            }
        }
    }
}

// ---------------------------------------------------------------------------
// energy_H: per (b, w): E[h, g] = sum_c q[b,p(h,w),c] * k[b,p(g,w),c]   (f32)
// grid = (H/32, H/32, B*W), block = 32
// ---------------------------------------------------------------------------
__global__ __launch_bounds__(32)
void energy_h_kernel(const half_t* __restrict__ qh, const half_t* __restrict__ kk,
                     float* __restrict__ E)
{
    const int bw = blockIdx.z;
    const int b = bw / W_, w = bw % W_;
    const int m0 = blockIdx.x * 32;   // h tile
    const int n0 = blockIdx.y * 32;   // g tile
    const int lane = threadIdx.x;
    const int mrow = lane & 15, kh = lane >> 4;

    const half_t* qb = qh + (size_t)b * N_ * CQ_;
    const half_t* kb = kk + (size_t)b * N_ * CQ_;
    const half_t* ar0 = qb + (size_t)((m0 + mrow) * W_ + w) * CQ_;
    const half_t* ar1 = qb + (size_t)((m0 + 16 + mrow) * W_ + w) * CQ_;
    const half_t* bc0 = kb + (size_t)((n0 + mrow) * W_ + w) * CQ_;
    const half_t* bc1 = kb + (size_t)((n0 + 16 + mrow) * W_ + w) * CQ_;

    v8f acc[2][2] = {};
#pragma unroll
    for (int c0 = 0; c0 < CQ_; c0 += 32) {
        v16h a0 = load_a(ar0, c0, kh), a1 = load_a(ar1, c0, kh);
        v16h b0 = load_b(bc0, c0, kh), b1 = load_b(bc1, c0, kh);
        acc[0][0] = wmma_f16(a0, b0, acc[0][0]);
        acc[0][1] = wmma_f16(a0, b1, acc[0][1]);
        acc[1][0] = wmma_f16(a1, b0, acc[1][0]);
        acc[1][1] = wmma_f16(a1, b1, acc[1][1]);
    }

#pragma unroll
    for (int mi = 0; mi < 2; ++mi) {
#pragma unroll
        for (int i = 0; i < 8; ++i) {
            const int m = m0 + mi * 16 + 8 * kh + i;     // h index
            float* ep = E + ((size_t)(b * H_ + m) * W_ + w) * L_ + n0 + mrow;
            ep[0]  = acc[mi][0][i];
            ep[16] = acc[mi][1][i];
        }
    }
}

// ---------------------------------------------------------------------------
// energy_W: per (b, h): E[w, v] = sum_c q[b,p(h,w),c] * k[b,p(h,v),c]   (f32)
// grid = (W/32, W/32, B*H), block = 32
// ---------------------------------------------------------------------------
__global__ __launch_bounds__(32)
void energy_w_kernel(const half_t* __restrict__ qh, const half_t* __restrict__ kk,
                     float* __restrict__ E)
{
    const int bh = blockIdx.z;
    const int b = bh / H_, h = bh % H_;
    const int m0 = blockIdx.x * 32;   // w tile
    const int n0 = blockIdx.y * 32;   // v tile
    const int lane = threadIdx.x;
    const int mrow = lane & 15, kh = lane >> 4;

    const half_t* qb = qh + ((size_t)b * N_ + h * W_) * CQ_;
    const half_t* kb = kk + ((size_t)b * N_ + h * W_) * CQ_;
    const half_t* ar0 = qb + (size_t)(m0 + mrow) * CQ_;
    const half_t* ar1 = qb + (size_t)(m0 + 16 + mrow) * CQ_;
    const half_t* bc0 = kb + (size_t)(n0 + mrow) * CQ_;
    const half_t* bc1 = kb + (size_t)(n0 + 16 + mrow) * CQ_;

    v8f acc[2][2] = {};
#pragma unroll
    for (int c0 = 0; c0 < CQ_; c0 += 32) {
        v16h a0 = load_a(ar0, c0, kh), a1 = load_a(ar1, c0, kh);
        v16h b0 = load_b(bc0, c0, kh), b1 = load_b(bc1, c0, kh);
        acc[0][0] = wmma_f16(a0, b0, acc[0][0]);
        acc[0][1] = wmma_f16(a0, b1, acc[0][1]);
        acc[1][0] = wmma_f16(a1, b0, acc[1][0]);
        acc[1][1] = wmma_f16(a1, b1, acc[1][1]);
    }

#pragma unroll
    for (int mi = 0; mi < 2; ++mi) {
#pragma unroll
        for (int i = 0; i < 8; ++i) {
            const int m = m0 + mi * 16 + 8 * kh + i;     // w index
            float* ep = E + ((size_t)(b * H_ + h) * W_ + m) * L_ + 96 + n0 + mrow;
            ep[0]  = acc[mi][0][i];
            ep[16] = acc[mi][1][i];
        }
    }
}

// ---------------------------------------------------------------------------
// Softmax over 192-length rows: read f32 energies, write f16 attention.
// Diagonal mask (-inf at index h). One wave32 per pixel, 6 elements/lane.
// grid = B*N/8, block = 256
// ---------------------------------------------------------------------------
__global__ __launch_bounds__(256)
void softmax_kernel(const float* __restrict__ E, half_t* __restrict__ att)
{
    const int wave = threadIdx.x >> 5;
    const int lane = threadIdx.x & 31;
    const size_t pix = (size_t)blockIdx.x * 8 + wave;   // < B*H*W
    const int h = (int)((pix / W_) % H_);

    const float* e = E + pix * L_;
    half_t* ao = att + pix * L_;
    float v[6];
#pragma unroll
    for (int i = 0; i < 6; ++i) {
        const int idx = lane + 32 * i;
        v[i] = e[idx];
        if (idx == h) v[i] = -__builtin_inff();          // diag mask on energy_H
    }
    float m = v[0];
#pragma unroll
    for (int i = 1; i < 6; ++i) m = fmaxf(m, v[i]);
#pragma unroll
    for (int off = 16; off > 0; off >>= 1) m = fmaxf(m, __shfl_xor(m, off, 32));
    float s = 0.f;
#pragma unroll
    for (int i = 0; i < 6; ++i) { v[i] = __expf(v[i] - m); s += v[i]; }
#pragma unroll
    for (int off = 16; off > 0; off >>= 1) s += __shfl_xor(s, off, 32);
    const float inv = 1.0f / s;
#pragma unroll
    for (int i = 0; i < 6; ++i) ao[lane + 32 * i] = (half_t)(v[i] * inv);
}

// ---------------------------------------------------------------------------
// out_H: per (b, w): Out[c, h] = sum_g vtr[b,c,w,g] * att[b,h,w,g]
// written f32 into d_out (scratch for the H pass).
// 64x32 tile per wave (4x2 accumulators). grid = (C/64, H/32, B*W), block = 32
// ---------------------------------------------------------------------------
__global__ __launch_bounds__(32)
void out_h_kernel(const half_t* __restrict__ vtr, const half_t* __restrict__ att,
                  float* __restrict__ out)
{
    const int bw = blockIdx.z;
    const int b = bw / W_, w = bw % W_;
    const int m0 = blockIdx.x * 64;   // c tile
    const int n0 = blockIdx.y * 32;   // h tile
    const int lane = threadIdx.x;
    const int mrow = lane & 15, kh = lane >> 4;

    const half_t* ar[4];
#pragma unroll
    for (int mi = 0; mi < 4; ++mi)
        ar[mi] = vtr + ((size_t)(b * C_ + m0 + mi * 16 + mrow) * W_ + w) * H_;
    const half_t* bc[2];
#pragma unroll
    for (int ni = 0; ni < 2; ++ni)
        bc[ni] = att + ((size_t)(b * H_ + n0 + ni * 16 + mrow) * W_ + w) * L_;

    v8f acc[4][2] = {};
#pragma unroll
    for (int g0 = 0; g0 < H_; g0 += 32) {
        v16h a[4], bv[2];
#pragma unroll
        for (int mi = 0; mi < 4; ++mi) a[mi] = load_a(ar[mi], g0, kh);
#pragma unroll
        for (int ni = 0; ni < 2; ++ni) bv[ni] = load_b(bc[ni], g0, kh);
#pragma unroll
        for (int mi = 0; mi < 4; ++mi)
#pragma unroll
            for (int ni = 0; ni < 2; ++ni)
                acc[mi][ni] = wmma_f16(a[mi], bv[ni], acc[mi][ni]);
    }

#pragma unroll
    for (int mi = 0; mi < 4; ++mi) {
#pragma unroll
        for (int i = 0; i < 8; ++i) {
            const int m = m0 + mi * 16 + 8 * kh + i;     // channel
            float* op = out + ((size_t)(b * C_ + m) * H_ + n0 + mrow) * W_ + w;
            op[0]       = acc[mi][0][i];
            op[16 * W_] = acc[mi][1][i];
        }
    }
}

// ---------------------------------------------------------------------------
// out_W + epilogue: per (b, h): OutW[c, w] = sum_v vnat[b,c,h,v]*att[b,h,w,96+v]
// final: out = gamma * (out_H(scratch in d_out) + OutW) + x
// 64x32 tile per wave. grid = (C/64, W/32, B*H), block = 32
// ---------------------------------------------------------------------------
__global__ __launch_bounds__(32)
void out_w_kernel(const half_t* __restrict__ vnat, const half_t* __restrict__ att,
                  const float* __restrict__ x, const float* __restrict__ gamma,
                  float* __restrict__ out)
{
    const int bh = blockIdx.z;
    const int b = bh / H_, h = bh % H_;
    const int m0 = blockIdx.x * 64;   // c tile
    const int n0 = blockIdx.y * 32;   // w tile
    const int lane = threadIdx.x;
    const int mrow = lane & 15, kh = lane >> 4;

    const half_t* ar[4];
#pragma unroll
    for (int mi = 0; mi < 4; ++mi)
        ar[mi] = vnat + ((size_t)(b * C_ + m0 + mi * 16 + mrow) * H_ + h) * W_;
    const half_t* bc[2];
#pragma unroll
    for (int ni = 0; ni < 2; ++ni)
        bc[ni] = att + ((size_t)(b * H_ + h) * W_ + n0 + ni * 16 + mrow) * L_ + 96;

    v8f acc[4][2] = {};
#pragma unroll
    for (int g0 = 0; g0 < W_; g0 += 32) {
        v16h a[4], bv[2];
#pragma unroll
        for (int mi = 0; mi < 4; ++mi) a[mi] = load_a(ar[mi], g0, kh);
#pragma unroll
        for (int ni = 0; ni < 2; ++ni) bv[ni] = load_b(bc[ni], g0, kh);
#pragma unroll
        for (int mi = 0; mi < 4; ++mi)
#pragma unroll
            for (int ni = 0; ni < 2; ++ni)
                acc[mi][ni] = wmma_f16(a[mi], bv[ni], acc[mi][ni]);
    }

    const float g0v = gamma[0];
#pragma unroll
    for (int mi = 0; mi < 4; ++mi) {
#pragma unroll
        for (int i = 0; i < 8; ++i) {
            const int m = m0 + mi * 16 + 8 * kh + i;     // channel
            const size_t base = ((size_t)(b * C_ + m) * H_ + h) * W_ + n0 + mrow;
            out[base]      = g0v * (out[base]      + acc[mi][0][i]) + x[base];
            out[base + 16] = g0v * (out[base + 16] + acc[mi][1][i]) + x[base + 16];
        }
    }
}

// ---------------------------------------------------------------------------
extern "C" void kernel_launch(void* const* d_in, const int* in_sizes, int n_in,
                              void* d_out, int out_size, void* d_ws, size_t ws_size,
                              hipStream_t stream) {
    (void)in_sizes; (void)n_in; (void)out_size; (void)ws_size;

    const float* x     = (const float*)d_in[0];
    const float* y     = (const float*)d_in[1];
    const float* Wq    = (const float*)d_in[2];
    const float* bq    = (const float*)d_in[3];
    const float* Wk    = (const float*)d_in[4];
    const float* bk    = (const float*)d_in[5];
    const float* Wv    = (const float*)d_in[6];
    const float* bv    = (const float*)d_in[7];
    const float* gamma = (const float*)d_in[8];
    float* out = (float*)d_out;

    // Workspace carve-up (f16 region first, f32 energies at the end)
    const size_t SZ_XN  = (size_t)B_ * N_ * C_;    // 37,748,736 halfs
    const size_t SZ_QK  = (size_t)B_ * N_ * CQ_;   //  4,718,592 halfs
    const size_t SZ_ATT = (size_t)B_ * N_ * L_;    // 14,155,776 halfs

    half_t* hws  = (half_t*)d_ws;
    half_t* xh   = hws;
    half_t* yh   = xh   + SZ_XN;
    half_t* vnat = yh   + SZ_XN;
    half_t* vtr  = vnat + SZ_XN;
    half_t* qh   = vtr  + SZ_XN;
    half_t* kh   = qh   + SZ_QK;
    half_t* atth = kh   + SZ_QK;
    half_t* wqh  = atth + SZ_ATT;
    half_t* wkh  = wqh  + (size_t)CQ_ * C_;
    half_t* wvh  = wkh  + (size_t)CQ_ * C_;
    float*  Eb   = (float*)(wvh + (size_t)C_ * C_);

    // 1) Weight conversion f32 -> f16
    f32_to_f16_kernel<<<dim3((CQ_ * C_) / 256), 256, 0, stream>>>(Wq, wqh, CQ_ * C_);
    f32_to_f16_kernel<<<dim3((CQ_ * C_) / 256), 256, 0, stream>>>(Wk, wkh, CQ_ * C_);
    f32_to_f16_kernel<<<dim3((C_  * C_) / 256), 256, 0, stream>>>(Wv, wvh, C_ * C_);

    // 2) Input transpose to pixel-major f16
    transpose_to_f16_kernel<<<dim3(N_ / 32, C_ / 32, B_), dim3(32, 8), 0, stream>>>(x, xh);
    transpose_to_f16_kernel<<<dim3(N_ / 32, C_ / 32, B_), dim3(32, 8), 0, stream>>>(y, yh);

    // 3) Projections (q, k pixel-major f16; v dual-layout f16), 64x64 tiles
    proj_qk_kernel<<<dim3(CQ_ / 64, N_ / 64, B_), 32, 0, stream>>>(wqh, bq, xh, qh, CQ_);
    proj_qk_kernel<<<dim3(CQ_ / 64, N_ / 64, B_), 32, 0, stream>>>(wkh, bk, yh, kh, CQ_);
    proj_v_kernel <<<dim3(C_  / 64, N_ / 64, B_), 32, 0, stream>>>(wvh, bv, xh, vnat, vtr);

    // 4) Energies (f32)
    energy_h_kernel<<<dim3(H_ / 32, H_ / 32, B_ * W_), 32, 0, stream>>>(qh, kh, Eb);
    energy_w_kernel<<<dim3(W_ / 32, W_ / 32, B_ * H_), 32, 0, stream>>>(qh, kh, Eb);

    // 5) Softmax with diag mask -> f16 attention
    softmax_kernel<<<dim3((B_ * N_) / 8), 256, 0, stream>>>(Eb, atth);

    // 6) Aggregation: H pass into d_out scratch, then W pass + epilogue
    out_h_kernel<<<dim3(C_ / 64, H_ / 32, B_ * W_), 32, 0, stream>>>(vtr, atth, out);
    out_w_kernel<<<dim3(C_ / 64, W_ / 32, B_ * H_), 32, 0, stream>>>(vnat, atth, x, gamma, out);
}
